// LocalCrossAttentionFusion_28338194219149
// MI455X (gfx1250) — matrix operand order repaired and checked by reading, not compiled, not run on hardware
//
#include <hip/hip_runtime.h>
#include <hip/hip_fp16.h>

typedef _Float16 half_t;
typedef __attribute__((ext_vector_type(16))) _Float16 v16h;
typedef __attribute__((ext_vector_type(8)))  _Float16 v8h;
typedef __attribute__((ext_vector_type(8)))  float    v8f;
typedef __attribute__((ext_vector_type(4)))  unsigned int u32x4;
typedef __attribute__((ext_vector_type(8)))  int i32x8;
typedef __attribute__((ext_vector_type(4)))  int i32x4;

#define BB   2
#define TT   2048
#define SS   2048
#define DD   512
#define HH   8
#define HDD  64
#define WWIN 65
#define WLQ  32

// LDS A-tile: 32 rows x 512 f16, row stride padded 1024B -> 1040B (260 dwords, bank-spread)
#define LDS_ROW_BYTES 1040
#define MTILE 32

#if defined(__gfx1250__) && __has_builtin(__builtin_amdgcn_tensor_load_to_lds)
#define USE_TDM 1
#endif

// ---------------------------------------------------------------- cast f32->f16
__global__ void cast_f32_f16(const float* __restrict__ src, half_t* __restrict__ dst, int n) {
    int i = blockIdx.x * blockDim.x + threadIdx.x;
    if (i < n) dst[i] = (half_t)src[i];
}

// ---------------------------------------------------------------- row LayerNorm -> f16
__global__ __launch_bounds__(128) void ln_rows_f16(const float* __restrict__ x,
                                                   const float* __restrict__ gamma,
                                                   const float* __restrict__ beta,
                                                   half_t* __restrict__ out) {
    __shared__ float s_sum[4];
    __shared__ float s_var[4];
    const int row  = blockIdx.x;
    const int tid  = threadIdx.x;
    const int lane = tid & 31;
    const int wid  = tid >> 5;
    const float* xr = x + (size_t)row * DD;
    float v[4];
    float s = 0.f;
#pragma unroll
    for (int i = 0; i < 4; ++i) { v[i] = xr[tid + 128 * i]; s += v[i]; }
#pragma unroll
    for (int m = 16; m >= 1; m >>= 1) s += __shfl_xor(s, m, 32);
    if (lane == 0) s_sum[wid] = s;
    __syncthreads();
    const float mu = (s_sum[0] + s_sum[1] + s_sum[2] + s_sum[3]) * (1.f / DD);
    float vs = 0.f;
#pragma unroll
    for (int i = 0; i < 4; ++i) { float d = v[i] - mu; vs += d * d; }
#pragma unroll
    for (int m = 16; m >= 1; m >>= 1) vs += __shfl_xor(vs, m, 32);
    if (lane == 0) s_var[wid] = vs;
    __syncthreads();
    const float var = (s_var[0] + s_var[1] + s_var[2] + s_var[3]) * (1.f / DD);
    const float rs  = rsqrtf(var + 1e-5f);
    half_t* orow = out + (size_t)row * DD;
#pragma unroll
    for (int i = 0; i < 4; ++i) {
        int c = tid + 128 * i;
        orow[c] = (half_t)((v[i] - mu) * rs * gamma[c] + beta[c]);
    }
}

// ---------------------------------------------------------------- fragment helpers
__device__ __forceinline__ v16h load_bfrag(const half_t* __restrict__ p) {
    // B 32x16 f16, ISA layout: N = lane%16; lanes0-15 K=0..7,16..23; lanes16-31 K=8..15,24..31 (8*hs pre-added)
    v8h l0 = *(const v8h*)(p);
    v8h l1 = *(const v8h*)(p + 16);
    v16h r;
#pragma unroll
    for (int i = 0; i < 8; ++i) { r[i] = l0[i]; r[i + 8] = l1[i]; }
    return r;
}

__device__ __forceinline__ v16h lds_afrag(const char* lds_base, int row, int kk, int hs) {
    // A 16x32 f16, ISA layout: lane row = lane%16; lanes0-15 K=0..7,16..23; lanes16-31 K=8..15,24..31
    const char* rp = lds_base + row * LDS_ROW_BYTES;
    v8h lo = *(const v8h*)(rp + (kk + 8 * hs) * 2);
    v8h hi = *(const v8h*)(rp + (kk + 16 + 8 * hs) * 2);
    v16h r;
#pragma unroll
    for (int i = 0; i < 8; ++i) { r[i] = lo[i]; r[i + 8] = hi[i]; }
    return r;
}

__device__ __forceinline__ void wmma_batch(v8f (&acc0)[4], v8f (&acc1)[4],
                                           const v16h& a0, const v16h& a1,
                                           const v16h (&b)[4]) {
#pragma unroll
    for (int c = 0; c < 4; ++c)
        acc0[c] = __builtin_amdgcn_wmma_f32_16x16x32_f16(
            false, a0, false, b[c], (short)0, acc0[c], false, false);
#pragma unroll
    for (int c = 0; c < 4; ++c)
        acc1[c] = __builtin_amdgcn_wmma_f32_16x16x32_f16(
            false, a1, false, b[c], (short)0, acc1[c], false, false);
}

// ---------------------------------------------------------------- GEMM: C = A(4096x512,f16) @ Bw^T + bias (+resid)
// Block covers 32 rows x 512 cols; 8 waves each own a 32x64 tile. Shared 32x512 A-tile
// staged in LDS via the Tensor Data Mover (one tensor_load_to_lds per block).
__global__ __launch_bounds__(256) void gemm_wmma_512(const half_t* __restrict__ A,
                                                     const half_t* __restrict__ Bw,
                                                     const float*  __restrict__ bias,
                                                     const float*  __restrict__ resid,
                                                     float* __restrict__ C) {
    __shared__ uint4 lds_a[MTILE * (LDS_ROW_BYTES / 16)];
    const int lane   = threadIdx.x & 31;
    const int wave   = threadIdx.x >> 5;      // 0..7
    const int hs     = lane >> 4;             // lane half (0/1)
    const int lr     = lane & 15;
    const int m_base = blockIdx.x * MTILE;
    const int n_base = wave * 64;

#if defined(USE_TDM)
    if (threadIdx.x == 0) {
        const unsigned lds_off = (unsigned)(unsigned long long)(void*)&lds_a[0];
        const unsigned long long ga =
            (unsigned long long)(const void*)(A + (size_t)m_base * DD);
        // D# group 0: count=1, lds_addr, global_addr[56:0], type=2 (bits 127:126 = 10b)
        u32x4 g0 = { 1u, lds_off, (unsigned)ga,
                     (unsigned)((ga >> 32) & 0x01FFFFFFull) | 0x80000000u };
        // D# group 1: data_size=2B(code1), pad_enable, pad_interval=256dw(code7),
        // pad_amount=4dw(code3); tensor 512 x 4096 (stride0=512); tile 512 x 32.
        i32x8 g1 = { (int)((1u << 16) | (1u << 20) | (7u << 22) | (3u << 25)),
                     (int)(512u << 16),            // tensor_dim0[15:0]=512
                     (int)(4096u << 16),           // tensor_dim0[31:16]=0, tensor_dim1[15:0]=4096
                     (int)(512u << 16),            // tensor_dim1[31:16]=0, tile_dim0=512
                     (int)32u,                     // tile_dim1=32, tile_dim2=0
                     (int)512u,                    // tensor_dim0_stride[31:0]=512
                     0, 0 };
        i32x4 g2 = { 0, 0, 0, 0 };
        i32x4 g3 = { 0, 0, 0, 0 };
#if __clang_major__ >= 23
        i32x8 g4 = { 0, 0, 0, 0, 0, 0, 0, 0 };
        __builtin_amdgcn_tensor_load_to_lds(g0, g1, g2, g3, g4, 0);
#else
        __builtin_amdgcn_tensor_load_to_lds(g0, g1, g2, g3, 0);
#endif
        __builtin_amdgcn_s_wait_tensorcnt(0);
    }
#else
    // fallback: cooperative global->LDS staging of the padded A tile
    for (int i = threadIdx.x; i < MTILE * 64; i += 256) {
        const int r = i >> 6, c = i & 63;
        const uint4 src = *((const uint4*)(A + (size_t)(m_base + r) * DD) + c);
        *(uint4*)((char*)lds_a + r * LDS_ROW_BYTES + c * 16) = src;
    }
#endif
    __syncthreads();
    const char* lbase = (const char*)lds_a;

    v8f acc0[4] = {v8f{}, v8f{}, v8f{}, v8f{}};
    v8f acc1[4] = {v8f{}, v8f{}, v8f{}, v8f{}};
    const half_t* brow[4];
#pragma unroll
    for (int c = 0; c < 4; ++c)
        brow[c] = Bw + (size_t)(n_base + 16 * c + lr) * DD + 8 * hs;

    // double-buffered B fragments (no rotate copies -> no WMMA hazard NOPs)
    v16h b0[4], b1[4];
#pragma unroll
    for (int c = 0; c < 4; ++c) b0[c] = load_bfrag(brow[c]);

    for (int kk = 0; kk < DD - 64; kk += 64) {
#pragma unroll
        for (int c = 0; c < 4; ++c) b1[c] = load_bfrag(brow[c] + kk + 32);
        const v16h a0 = lds_afrag(lbase, lr,      kk, hs);
        const v16h a1 = lds_afrag(lbase, lr + 16, kk, hs);
        const v16h a2 = lds_afrag(lbase, lr,      kk + 32, hs);
        const v16h a3 = lds_afrag(lbase, lr + 16, kk + 32, hs);
        wmma_batch(acc0, acc1, a0, a1, b0);
#pragma unroll
        for (int c = 0; c < 4; ++c) b0[c] = load_bfrag(brow[c] + kk + 64);
        wmma_batch(acc0, acc1, a2, a3, b1);
    }
    {   // epilogue: kk = DD-64 (b0 already holds frag(DD-64))
        const int kk = DD - 64;
#pragma unroll
        for (int c = 0; c < 4; ++c) b1[c] = load_bfrag(brow[c] + kk + 32);
        const v16h a0 = lds_afrag(lbase, lr,      kk, hs);
        const v16h a1 = lds_afrag(lbase, lr + 16, kk, hs);
        const v16h a2 = lds_afrag(lbase, lr,      kk + 32, hs);
        const v16h a3 = lds_afrag(lbase, lr + 16, kk + 32, hs);
        wmma_batch(acc0, acc1, a0, a1, b0);
        wmma_batch(acc0, acc1, a2, a3, b1);
    }

#pragma unroll
    for (int c = 0; c < 4; ++c) {
        const int n = n_base + 16 * c + lr;
        const float bb = bias ? bias[n] : 0.f;
#pragma unroll
        for (int r = 0; r < 8; ++r) {
            const int m0 = m_base + r + 8 * hs;          // group 0: rows 0..15
            const int m1 = m0 + 16;                      // group 1: rows 16..31
            const size_t off0 = (size_t)m0 * DD + n;
            const size_t off1 = (size_t)m1 * DD + n;
            float v0 = acc0[c][r] + bb;
            float v1 = acc1[c][r] + bb;
            if (resid) { v0 += resid[off0]; v1 += resid[off1]; }
            C[off0] = v0;
            C[off1] = v1;
        }
    }
}

// ---------------------------------------------------------------- local windowed attention
__global__ __launch_bounds__(256) void attn_local(const float* __restrict__ Qf,
                                                  const float* __restrict__ Kf,
                                                  const float* __restrict__ Vf,
                                                  const int* __restrict__ qpos,
                                                  const int* __restrict__ clens,
                                                  half_t* __restrict__ Ff) {
    __shared__ float sq[8][HDD];
    __shared__ float sp[8][WWIN];
    __shared__ int   srow[8][WWIN];
    const int lane = threadIdx.x & 31;
    const int wid  = threadIdx.x >> 5;
    const int gid  = blockIdx.x * 8 + wid;      // (b,t,h)
    const int h = gid % HH;
    const int t = (gid / HH) % TT;
    const int b = gid / (HH * TT);
    const int pos  = qpos[b * TT + t];
    const int clen = clens[b];

    const float* qrow = Qf + ((size_t)(b * TT + t) * DD + h * HDD);
    sq[wid][lane]      = qrow[lane];
    sq[wid][lane + 32] = qrow[lane + 32];
    __syncthreads();

    const float NEGF = -3.402823466e38f;  // finfo(f32).min
    float sc[3]; int rofs[3]; int vld[3];
#pragma unroll
    for (int j = 0; j < 3; ++j) {
        const int w = lane + 32 * j;
        sc[j] = NEGF; vld[j] = 0; rofs[j] = 0;
        if (w < WWIN) {
            const int idx   = pos - WLQ + w;
            const int valid = (idx >= 0) && (idx < clen);
            const int ci    = idx < 0 ? 0 : (idx > SS - 1 ? SS - 1 : idx);
            const int ro    = (b * SS + ci) * DD + h * HDD;
            const float4* kp = (const float4*)(Kf + ro);
            float d = 0.f;
#pragma unroll
            for (int i = 0; i < 16; ++i) {
                float4 kv = kp[i];
                d += sq[wid][4 * i + 0] * kv.x + sq[wid][4 * i + 1] * kv.y
                   + sq[wid][4 * i + 2] * kv.z + sq[wid][4 * i + 3] * kv.w;
            }
            sc[j]   = valid ? d * 0.125f : NEGF;   // 1/sqrt(64)
            vld[j]  = valid;
            rofs[j] = ro;
        }
    }
    // masked softmax (reference semantics) + renormalize
    float m = fmaxf(fmaxf(sc[0], sc[1]), sc[2]);
#pragma unroll
    for (int mm = 16; mm >= 1; mm >>= 1) m = fmaxf(m, __shfl_xor(m, mm, 32));
    float e[3]; float ssum = 0.f;
#pragma unroll
    for (int j = 0; j < 3; ++j) { e[j] = vld[j] ? __expf(sc[j] - m) : 0.f; ssum += e[j]; }
#pragma unroll
    for (int mm = 16; mm >= 1; mm >>= 1) ssum += __shfl_xor(ssum, mm, 32);
    const float inv = 1.f / fmaxf(ssum, 1e-6f);
#pragma unroll
    for (int j = 0; j < 3; ++j) {
        const int w = lane + 32 * j;
        if (w < WWIN) { sp[wid][w] = e[j] * inv; srow[wid][w] = rofs[j]; }
    }
    __syncthreads();

    // fused = attn . V  (lanes own hd and hd+32 -> coalesced V reads)
    float a0 = 0.f, a1 = 0.f;
    for (int w = 0; w < WWIN; ++w) {
        const float p = sp[wid][w];
        const int   ro = srow[wid][w];
        a0 += p * Vf[ro + lane];
        a1 += p * Vf[ro + lane + 32];
    }
    half_t* frow = Ff + ((size_t)(b * TT + t) * DD + h * HDD);
    frow[lane]      = (half_t)a0;
    frow[lane + 32] = (half_t)a1;
}

// ---------------------------------------------------------------- launcher
extern "C" void kernel_launch(void* const* d_in, const int* in_sizes, int n_in,
                              void* d_out, int out_size, void* d_ws, size_t ws_size,
                              hipStream_t stream) {
    const float* query   = (const float*)d_in[0];
    const float* context = (const float*)d_in[1];
    const int*   qpos    = (const int*)d_in[2];
    const int*   clens   = (const int*)d_in[3];
    const float* ln_q_g  = (const float*)d_in[4];
    const float* ln_q_b  = (const float*)d_in[5];
    const float* ln_c_g  = (const float*)d_in[6];
    const float* ln_c_b  = (const float*)d_in[7];
    const float* Wq = (const float*)d_in[8];
    const float* bq = (const float*)d_in[9];
    const float* Wk = (const float*)d_in[10];
    const float* bk = (const float*)d_in[11];
    const float* Wv = (const float*)d_in[12];
    const float* bv = (const float*)d_in[13];
    const float* Wo = (const float*)d_in[14];
    const float* bo = (const float*)d_in[15];
    float* out = (float*)d_out;
    (void)in_sizes; (void)n_in; (void)out_size; (void)ws_size;

    const int NROW = BB * TT;   // 4096 rows for both query and context batches
    char* ws = (char*)d_ws;
    size_t off = 0;
    auto alloc = [&](size_t bytes) -> void* {
        void* p = ws + off;
        off += (bytes + 255) & ~(size_t)255;
        return p;
    };
    half_t* Xq  = (half_t*)alloc((size_t)NROW * DD * sizeof(half_t));
    half_t* Xc  = (half_t*)alloc((size_t)NROW * DD * sizeof(half_t));
    half_t* Wqh = (half_t*)alloc((size_t)DD * DD * sizeof(half_t));
    half_t* Wkh = (half_t*)alloc((size_t)DD * DD * sizeof(half_t));
    half_t* Wvh = (half_t*)alloc((size_t)DD * DD * sizeof(half_t));
    half_t* Woh = (half_t*)alloc((size_t)DD * DD * sizeof(half_t));
    float*  Qf  = (float*)alloc((size_t)NROW * DD * sizeof(float));
    float*  Kf  = (float*)alloc((size_t)NROW * DD * sizeof(float));
    float*  Vf  = (float*)alloc((size_t)NROW * DD * sizeof(float));
    half_t* Ff  = (half_t*)alloc((size_t)NROW * DD * sizeof(half_t));

    const int nw = DD * DD;
    cast_f32_f16<<<(nw + 255) / 256, 256, 0, stream>>>(Wq, Wqh, nw);
    cast_f32_f16<<<(nw + 255) / 256, 256, 0, stream>>>(Wk, Wkh, nw);
    cast_f32_f16<<<(nw + 255) / 256, 256, 0, stream>>>(Wv, Wvh, nw);
    cast_f32_f16<<<(nw + 255) / 256, 256, 0, stream>>>(Wo, Woh, nw);

    ln_rows_f16<<<NROW, 128, 0, stream>>>(query,   ln_q_g, ln_q_b, Xq);
    ln_rows_f16<<<NROW, 128, 0, stream>>>(context, ln_c_g, ln_c_b, Xc);

    gemm_wmma_512<<<NROW / MTILE, 256, 0, stream>>>(Xq, Wqh, bq, nullptr, Qf);
    gemm_wmma_512<<<NROW / MTILE, 256, 0, stream>>>(Xc, Wkh, bk, nullptr, Kf);
    gemm_wmma_512<<<NROW / MTILE, 256, 0, stream>>>(Xc, Wvh, bv, nullptr, Vf);

    attn_local<<<(BB * TT * HH) / 8, 256, 0, stream>>>(Qf, Kf, Vf, qpos, clens, Ff);

    gemm_wmma_512<<<NROW / MTILE, 256, 0, stream>>>(Ff, Woh, bo, query, out);
}
